// Indexer_82703890251805
// MI455X (gfx1250) — matrix operand order repaired and checked by reading, not compiled, not run on hardware
//
#include <hip/hip_runtime.h>
#include <hip/hip_bf16.h>

typedef __attribute__((ext_vector_type(16))) __bf16 v16bf;
typedef __attribute__((ext_vector_type(8)))  float  v8f;

#define S_LEN   4096
#define HIDDEN  2048
#define NT_REAL 21      // 16 q-tiles + 4 k-tiles + 1 w-tile (padded to 16 cols)
#define NT_PAD  24      // padded so every wave owns exactly 3 tiles
#define KSTEPS  64      // 2048 / 32

union Frag {
  uint4  u[2];
  v16bf  v;
};

__device__ __forceinline__ __bf16 f2bf(float x) {
  unsigned u = __builtin_bit_cast(unsigned, x);
  unsigned r = u + 0x7FFFu + ((u >> 16) & 1u);   // round-to-nearest-even
  unsigned short h = (unsigned short)(r >> 16);
  return __builtin_bit_cast(__bf16, h);
}

// ---------------------------------------------------------------------------
// Kernel 0: repack Wq/Wk/Ww (f32, row-major K x N) into bf16 WMMA B-fragment
// order: packed[((t*64 + s)*32 + lane)*16 + j], lane = (khalf<<4)|ncol,
// j -> K = s*32 + khalf*8 + (j<8 ? j : j+8). Tiles 21..23 are zero padding.
// ---------------------------------------------------------------------------
__global__ void pack_weights(const float* __restrict__ Wq,
                             const float* __restrict__ Wk,
                             const float* __restrict__ Ww,
                             __bf16* __restrict__ packed) {
  int tid = blockIdx.x * blockDim.x + threadIdx.x;
  if (tid >= NT_PAD * KSTEPS * 32) return;
  int lane = tid & 31;
  int s    = (tid >> 5) & 63;
  int t    = tid >> 11;
  int n    = lane & 15;
  int kh   = lane >> 4;
  __bf16* dst = packed + (size_t)tid * 16;
  int kbase = s * 32 + kh * 8;
#pragma unroll
  for (int j = 0; j < 16; ++j) {
    int k = kbase + (j < 8 ? j : j + 8);
    float v;
    if (t < 16)                  v = Wq[(size_t)k * 256 + t * 16 + n];
    else if (t < 20)             v = Wk[(size_t)k * 64 + (t - 16) * 16 + n];
    else if (t == 20 && n < 4)   v = Ww[(size_t)k * 4 + n];
    else                         v = 0.0f;
    dst[j] = f2bf(v);
  }
}

// ---------------------------------------------------------------------------
// Kernel 1: projection GEMM. hs (8192 x 2048) x packed weights -> qraw
// (8192x256), kraw (8192x64), wbuf (8192x4). One 16-row M-tile per block;
// wave wv owns tiles {wv, wv+8, wv+16} (statically 3 accumulators, no
// dynamic array indexing -> no scratch spills).
// ---------------------------------------------------------------------------
__global__ void __launch_bounds__(256)
proj_gemm(const float* __restrict__ hs,
          const __bf16* __restrict__ packedW,
          float* __restrict__ qraw,
          float* __restrict__ kraw,
          float* __restrict__ wbuf) {
  const int lane = threadIdx.x & 31;
  const int wv   = threadIdx.x >> 5;
  const int row0 = blockIdx.x * 16;
  const int m    = lane & 15;
  const int kh   = lane >> 4;

  const int t0 = wv;        // 0..7   : q-tile
  const int t1 = wv + 8;    // 8..15  : q-tile
  const int t2 = wv + 16;   // 16..23 : k-tile / w-tile / zero-dummy

  v8f zero8 = {0.f,0.f,0.f,0.f,0.f,0.f,0.f,0.f};
  v8f acc0 = zero8, acc1 = zero8, acc2 = zero8;

  // A: lane m = M-row, kh selects K-half. 16-bit A layout:
  // j in [0,8): K = kh*8+j ; j in [8,16): K = 16 + kh*8 + (j-8)
  const float*  ap  = hs + (size_t)(row0 + m) * HIDDEN + kh * 8;
  const __bf16* bp0 = packedW + ((size_t)t0 * KSTEPS * 32 + lane) * 16;
  const __bf16* bp1 = packedW + ((size_t)t1 * KSTEPS * 32 + lane) * 16;
  const __bf16* bp2 = packedW + ((size_t)t2 * KSTEPS * 32 + lane) * 16;

  for (int s = 0; s < KSTEPS; ++s) {
    float4 f0 = *(const float4*)(ap);
    float4 f1 = *(const float4*)(ap + 4);
    float4 f2 = *(const float4*)(ap + 16);
    float4 f3 = *(const float4*)(ap + 20);
    v16bf a;
    a[0]=f2bf(f0.x);  a[1]=f2bf(f0.y);  a[2]=f2bf(f0.z);  a[3]=f2bf(f0.w);
    a[4]=f2bf(f1.x);  a[5]=f2bf(f1.y);  a[6]=f2bf(f1.z);  a[7]=f2bf(f1.w);
    a[8]=f2bf(f2.x);  a[9]=f2bf(f2.y);  a[10]=f2bf(f2.z); a[11]=f2bf(f2.w);
    a[12]=f2bf(f3.x); a[13]=f2bf(f3.y); a[14]=f2bf(f3.z); a[15]=f2bf(f3.w);

    Frag b0, b1, b2;
    b0.u[0] = ((const uint4*)bp0)[0];  b0.u[1] = ((const uint4*)bp0)[1];
    b1.u[0] = ((const uint4*)bp1)[0];  b1.u[1] = ((const uint4*)bp1)[1];
    b2.u[0] = ((const uint4*)bp2)[0];  b2.u[1] = ((const uint4*)bp2)[1];

    acc0 = __builtin_amdgcn_wmma_f32_16x16x32_bf16(
        false, a, false, b0.v, (short)0, acc0, false, false);
    acc1 = __builtin_amdgcn_wmma_f32_16x16x32_bf16(
        false, a, false, b1.v, (short)0, acc1, false, false);
    acc2 = __builtin_amdgcn_wmma_f32_16x16x32_bf16(
        false, a, false, b2.v, (short)0, acc2, false, false);

    ap  += 32;          // next 32-K step
    bp0 += 32 * 16;     // per-step fragment stride (32 lanes * 16 halves)
    bp1 += 32 * 16;
    bp2 += 32 * 16;
  }

  // C layout: VGPR v -> row = v + 8*kh, col = lane&15
#pragma unroll
  for (int v = 0; v < 8; ++v) {
    int grow = row0 + v + 8 * kh;
    qraw[(size_t)grow * 256 + t0 * 16 + m] = acc0[v];
    qraw[(size_t)grow * 256 + t1 * 16 + m] = acc1[v];
    if (t2 < 20)          kraw[(size_t)grow * 64 + (t2 - 16) * 16 + m] = acc2[v];
    else if (t2 == 20) {  if (m < 4) wbuf[(size_t)grow * 4 + m] = acc2[v]; }
  }
}

// ---------------------------------------------------------------------------
// Kernel 2: LayerNorm(k) + RoPE(q_pe, k_pe) + bf16 pack. One wave per row
// (k) or per (row, head) (q). wave32 shuffle reductions.
// ---------------------------------------------------------------------------
__global__ void __launch_bounds__(256)
ln_rope_pack(const float* __restrict__ qraw,
             const float* __restrict__ kraw,
             const float* __restrict__ cosb,
             const float* __restrict__ sinb,
             const float* __restrict__ gam,
             const float* __restrict__ bet,
             __bf16* __restrict__ qpk,
             __bf16* __restrict__ kpk) {
  int g    = blockIdx.x * 8 + (threadIdx.x >> 5);
  int lane = threadIdx.x & 31;
  const int NQ = 2 * S_LEN * 4;   // 32768 q-tasks
  if (g < NQ) {
    int row = g >> 2, h = g & 3;
    const float* qp = qraw + (size_t)row * 256 + h * 64;
    float v1 = qp[lane];           // lane 0..31 = rotary region
    float v2 = qp[lane + 32];
    float partner = qp[lane ^ 16];
    float c  = cosb[(size_t)row * 32 + lane];
    float sn = sinb[(size_t)row * 32 + lane];
    float rot = (lane < 16) ? -partner : partner;
    __bf16* o = qpk + (size_t)g * 64;
    o[lane]      = f2bf(v1 * c + rot * sn);
    o[lane + 32] = f2bf(v2);
  } else {
    int row = g - NQ;
    if (row >= 2 * S_LEN) return;
    const float* kp = kraw + (size_t)row * 64;
    float v1 = kp[lane], v2 = kp[lane + 32];
    float sum = v1 + v2;
#pragma unroll
    for (int msk = 16; msk >= 1; msk >>= 1) sum += __shfl_xor(sum, msk, 32);
    float mu = sum * (1.f / 64.f);
    float d1 = v1 - mu, d2 = v2 - mu;
    float vs = d1 * d1 + d2 * d2;
#pragma unroll
    for (int msk = 16; msk >= 1; msk >>= 1) vs += __shfl_xor(vs, msk, 32);
    float rstd = rsqrtf(vs * (1.f / 64.f) + 1e-5f);
    float n1 = d1 * rstd * gam[lane]      + bet[lane];
    float n2 = d2 * rstd * gam[lane + 32] + bet[lane + 32];
    float partner = __shfl_xor(n1, 16, 32);
    float c  = cosb[(size_t)row * 32 + lane];
    float sn = sinb[(size_t)row * 32 + lane];
    float rot = (lane < 16) ? -partner : partner;
    __bf16* o = kpk + (size_t)row * 64;
    o[lane]      = f2bf(n1 * c + rot * sn);
    o[lane + 32] = f2bf(n2);
  }
}

// ---------------------------------------------------------------------------
// Kernel 3: score = relu(q . k^T) per head, head-weighted sum, 128x128 tile
// per block; wave = 64x32 patch (4x2 WMMA C-tiles), K=64 in 2 steps of 32.
// ---------------------------------------------------------------------------
__global__ void __launch_bounds__(256)
score_kernel(const __bf16* __restrict__ qpk,   // (B*S, 4, 64)
             const __bf16* __restrict__ kpk,   // (B*S, 64)
             const float*  __restrict__ wbuf,  // (B*S, 4)
             float* __restrict__ out) {        // (B, S, S)
  __shared__ float ldsw[128 * 4];
  const int lane = threadIdx.x & 31;
  const int wv   = threadIdx.x >> 5;
  const int b    = blockIdx.z;
  const int q0   = blockIdx.y * 128;
  const int k0   = blockIdx.x * 128;
  const int m    = lane & 15;
  const int kh   = lane >> 4;

  for (int i = threadIdx.x; i < 512; i += 256)
    ldsw[i] = wbuf[((size_t)b * S_LEN + q0) * 4 + i];
  __syncthreads();

  const int QW = q0 + 64 * (wv >> 2);
  const int KW = k0 + 32 * (wv & 3);

  v8f zero8 = {0.f,0.f,0.f,0.f,0.f,0.f,0.f,0.f};
  v8f facc[4][2];
#pragma unroll
  for (int mi = 0; mi < 4; ++mi)
#pragma unroll
    for (int ni = 0; ni < 2; ++ni) facc[mi][ni] = zero8;

  for (int h = 0; h < 4; ++h) {
    v8f hacc[4][2];
#pragma unroll
    for (int mi = 0; mi < 4; ++mi)
#pragma unroll
      for (int ni = 0; ni < 2; ++ni) hacc[mi][ni] = zero8;

#pragma unroll
    for (int ks = 0; ks < 2; ++ks) {
      Frag a[4];
#pragma unroll
      for (int mi = 0; mi < 4; ++mi) {
        const __bf16* ap = qpk +
            (((size_t)(b * S_LEN + QW + mi * 16 + m)) * 4 + h) * 64 +
            ks * 32 + kh * 8;
        a[mi].u[0] = *(const uint4*)(ap);
        a[mi].u[1] = *(const uint4*)(ap + 16);
      }
      Frag bb[2];
#pragma unroll
      for (int ni = 0; ni < 2; ++ni) {
        const __bf16* bp = kpk +
            ((size_t)(b * S_LEN + KW + ni * 16 + m)) * 64 +
            ks * 32 + kh * 8;
        bb[ni].u[0] = *(const uint4*)(bp);
        bb[ni].u[1] = *(const uint4*)(bp + 16);
      }
#pragma unroll
      for (int mi = 0; mi < 4; ++mi)
#pragma unroll
        for (int ni = 0; ni < 2; ++ni)
          hacc[mi][ni] = __builtin_amdgcn_wmma_f32_16x16x32_bf16(
              false, a[mi].v, false, bb[ni].v, (short)0, hacc[mi][ni],
              false, false);
    }

    // relu + head weight, accumulate
#pragma unroll
    for (int mi = 0; mi < 4; ++mi)
#pragma unroll
      for (int v = 0; v < 8; ++v) {
        int lr = 64 * (wv >> 2) + mi * 16 + v + 8 * kh;
        float ww = ldsw[lr * 4 + h];
#pragma unroll
        for (int ni = 0; ni < 2; ++ni) {
          float x = hacc[mi][ni][v];
          x = x > 0.f ? x : 0.f;
          facc[mi][ni][v] += ww * x;
        }
      }
  }

#pragma unroll
  for (int mi = 0; mi < 4; ++mi)
#pragma unroll
    for (int ni = 0; ni < 2; ++ni)
#pragma unroll
      for (int v = 0; v < 8; ++v) {
        int qrow = QW + mi * 16 + v + 8 * kh;
        int kcol = KW + ni * 16 + m;
        out[((size_t)b * S_LEN + qrow) * (size_t)S_LEN + kcol] =
            facc[mi][ni][v];
      }
}

// ---------------------------------------------------------------------------
extern "C" void kernel_launch(void* const* d_in, const int* in_sizes, int n_in,
                              void* d_out, int out_size, void* d_ws, size_t ws_size,
                              hipStream_t stream) {
  const float* hs   = (const float*)d_in[0];
  const float* cosb = (const float*)d_in[1];
  const float* sinb = (const float*)d_in[2];
  const float* Wq   = (const float*)d_in[3];
  const float* Wk   = (const float*)d_in[4];
  const float* Ww   = (const float*)d_in[5];
  const float* gam  = (const float*)d_in[6];
  const float* bet  = (const float*)d_in[7];
  float* out = (float*)d_out;

  char* ws = (char*)d_ws;
  size_t off = 0;
  auto wsalloc = [&](size_t bytes) -> void* {
    void* p = ws + off;
    off = (off + bytes + 255) & ~(size_t)255;
    return p;
  };
  const size_t M = 2ull * S_LEN;                       // 8192 rows
  __bf16* packedW = (__bf16*)wsalloc((size_t)NT_PAD * KSTEPS * 32 * 16 * 2);
  float*  qraw    = (float*) wsalloc(M * 256 * 4);
  float*  kraw    = (float*) wsalloc(M * 64 * 4);
  float*  wbuf    = (float*) wsalloc(M * 4 * 4);
  __bf16* qpk     = (__bf16*)wsalloc(M * 256 * 2);
  __bf16* kpk     = (__bf16*)wsalloc(M * 64 * 2);

  pack_weights<<<(NT_PAD * KSTEPS * 32 + 255) / 256, 256, 0, stream>>>(
      Wq, Wk, Ww, packedW);
  proj_gemm<<<(int)(M / 16), 256, 0, stream>>>(hs, packedW, qraw, kraw, wbuf);
  ln_rope_pack<<<(int)((M * 4 + M) / 8), 256, 0, stream>>>(
      qraw, kraw, cosb, sinb, gam, bet, qpk, kpk);
  dim3 g2(S_LEN / 128, S_LEN / 128, 2);
  score_kernel<<<g2, 256, 0, stream>>>(qpk, kpk, wbuf, out);
}
